// NeRFGraph_20555713478895
// MI455X (gfx1250) — matrix operand order repaired and checked by reading, not compiled, not run on hardware
//
#include <hip/hip_runtime.h>
#include <hip/hip_bf16.h>
#include <math.h>

typedef _Float16 f16;
typedef __attribute__((ext_vector_type(16))) _Float16 v16h;
typedef __attribute__((ext_vector_type(8)))  _Float16 v8h;
typedef __attribute__((ext_vector_type(8)))  float    v8f;
typedef __attribute__((ext_vector_type(4)))  float    v4f;

#define NNODES 32768      // B*H*W = 2*128*128
#define HIMG 128
#define WIMG 128
#define HWIMG 16384
#define CIN 90            // CXYZ(63) + CDIR(27)
#define LDS_STRIDE 40     // halves per staged weight row (80B: 16B-aligned, bank-conflict-free)

// ---------------------------------------------------------------------------
// Fragment loaders (CDNA5 WMMA 16x16x32 f16 layouts, wave32)
// A (16x32, MxK): lane holds row M=lane%16; halves 0..7 = K {kb+hiK..+7},
//                 halves 8..15 = K {kb+16+hiK..+7}  (hiK = 8 for lanes>=16)
// B (32x16, KxN): lane holds col N=lane%16; same K order as A:
//                 halves 0..7 = K {kb+hiK..+7}, halves 8..15 = K {kb+16+hiK..+7}
// C/D (16x16 f32): lane holds col N=lane%16; vgpr i = row M=i (+8 if lane>=16)
// ---------------------------------------------------------------------------
__device__ __forceinline__ v16h loadA16(const f16* __restrict__ row, int kb, int hiK) {
  v8h lo = *(const v8h*)(row + kb + hiK);
  v8h hi = *(const v8h*)(row + kb + 16 + hiK);
  v16h a;
#pragma unroll
  for (int i = 0; i < 8; ++i) { a[i] = lo[i]; a[i + 8] = hi[i]; }
  return a;
}

__device__ __forceinline__ v16h loadB16(const f16* __restrict__ row, int kb, int hi16) {
  const f16* p = row + kb + (hi16 >> 1);
  v8h lo = *(const v8h*)(p);
  v8h hi = *(const v8h*)(p + 16);
  v16h b;
#pragma unroll
  for (int i = 0; i < 8; ++i) { b[i] = lo[i]; b[i + 8] = hi[i]; }
  return b;
}

// B-fragment from the LDS-staged weight tile (row-padded to LDS_STRIDE halves)
__device__ __forceinline__ v16h ldsFragB(const f16* lds, int row, int hi16) {
  const f16* p = lds + row * LDS_STRIDE + (hi16 >> 1);
  v8h lo = *(const v8h*)(p);
  v8h hi = *(const v8h*)(p + 16);
  v16h b;
#pragma unroll
  for (int i = 0; i < 8; ++i) { b[i] = lo[i]; b[i + 8] = hi[i]; }
  return b;
}

// Cooperative stage of a 64-row x 32-half weight tile into LDS (128 threads)
__device__ __forceinline__ void stageB(const f16* __restrict__ W, int ldW, int colBase, int kb,
                                       f16* __restrict__ lds, int tid) {
  const int row  = tid >> 1;            // 0..63
  const int half = (tid & 1) << 4;      // 0 or 16
  const f16* src = W + (colBase + row) * ldW + kb + half;
  v8h a = *(const v8h*)(src);
  v8h b = *(const v8h*)(src + 8);
  f16* dst = lds + row * LDS_STRIDE + half;
  *(v8h*)(dst)     = a;
  *(v8h*)(dst + 8) = b;
}

// ---------------------------------------------------------------------------
// Dual-operand WMMA GEMM: out = act( A1@W1^T + A2@W2^T + bias )
// 128-thread block = 4 waves; block tile 128 rows x 64 cols (wave: 32x64).
// Weight K-tile is double-buffered in LDS and shared by all 4 waves.
// Weight buffers zero-padded to (gridY*64) x (ksteps*32).
// ---------------------------------------------------------------------------
__global__ void __launch_bounds__(128) gemm2_wmma(
    const f16* __restrict__ A1, const f16* __restrict__ W1, int k1steps, int ldA1, int ldW1,
    const f16* __restrict__ A2, const f16* __restrict__ W2, int k2steps, int ldA2, int ldW2,
    const float* __restrict__ bias, int Mstore, int act,
    f16* __restrict__ outH, int ldH, float* __restrict__ outF, int ldF)
{
  __shared__ f16 ldsB[2][64 * LDS_STRIDE];

  const int tid   = threadIdx.x;
  const int lane  = tid & 31;
  const int wave  = tid >> 5;
  const int rlane = lane & 15;
  const int hiK   = (lane >> 4) << 3;   // 0 or 8
  const int hi16  = (lane >> 4) << 4;   // 0 or 16
  const int rowBase = blockIdx.x * 128 + wave * 32;
  const int colBase = blockIdx.y << 6;

  v8f c[2][4];
#pragma unroll
  for (int t = 0; t < 4; ++t) {
    const float bv = bias ? bias[colBase + t * 16 + rlane] : 0.0f;
#pragma unroll
    for (int i = 0; i < 8; ++i) { c[0][t][i] = bv; c[1][t][i] = bv; }
  }

  {
    const f16* arow0 = A1 + (rowBase + rlane) * ldA1;
    const f16* arow1 = A1 + (rowBase + 16 + rlane) * ldA1;
    stageB(W1, ldW1, colBase, 0, ldsB[0], tid);
    for (int ks = 0; ks < k1steps; ++ks) {
      const int kb = ks << 5;
      __syncthreads();
      if (ks + 1 < k1steps) stageB(W1, ldW1, colBase, (ks + 1) << 5, ldsB[(ks + 1) & 1], tid);
      __builtin_prefetch(arow0 + kb + 32, 0, 1);      // global_prefetch_b8 next A tile
      const f16* lb = ldsB[ks & 1];
      v16h b0 = ldsFragB(lb, 0  + rlane, hi16);
      v16h b1 = ldsFragB(lb, 16 + rlane, hi16);
      v16h b2 = ldsFragB(lb, 32 + rlane, hi16);
      v16h b3 = ldsFragB(lb, 48 + rlane, hi16);
      v16h a0 = loadA16(arow0, kb, hiK);
      v16h a1 = loadA16(arow1, kb, hiK);
      c[0][0] = __builtin_amdgcn_wmma_f32_16x16x32_f16(false, a0, false, b0, (short)0, c[0][0], false, false);
      c[0][1] = __builtin_amdgcn_wmma_f32_16x16x32_f16(false, a0, false, b1, (short)0, c[0][1], false, false);
      c[0][2] = __builtin_amdgcn_wmma_f32_16x16x32_f16(false, a0, false, b2, (short)0, c[0][2], false, false);
      c[0][3] = __builtin_amdgcn_wmma_f32_16x16x32_f16(false, a0, false, b3, (short)0, c[0][3], false, false);
      c[1][0] = __builtin_amdgcn_wmma_f32_16x16x32_f16(false, a1, false, b0, (short)0, c[1][0], false, false);
      c[1][1] = __builtin_amdgcn_wmma_f32_16x16x32_f16(false, a1, false, b1, (short)0, c[1][1], false, false);
      c[1][2] = __builtin_amdgcn_wmma_f32_16x16x32_f16(false, a1, false, b2, (short)0, c[1][2], false, false);
      c[1][3] = __builtin_amdgcn_wmma_f32_16x16x32_f16(false, a1, false, b3, (short)0, c[1][3], false, false);
    }
  }

  if (A2 != nullptr) {
    const f16* arow0 = A2 + (rowBase + rlane) * ldA2;
    const f16* arow1 = A2 + (rowBase + 16 + rlane) * ldA2;
    __syncthreads();
    stageB(W2, ldW2, colBase, 0, ldsB[0], tid);
    for (int ks = 0; ks < k2steps; ++ks) {
      const int kb = ks << 5;
      __syncthreads();
      if (ks + 1 < k2steps) stageB(W2, ldW2, colBase, (ks + 1) << 5, ldsB[(ks + 1) & 1], tid);
      const f16* lb = ldsB[ks & 1];
      v16h b0 = ldsFragB(lb, 0  + rlane, hi16);
      v16h b1 = ldsFragB(lb, 16 + rlane, hi16);
      v16h b2 = ldsFragB(lb, 32 + rlane, hi16);
      v16h b3 = ldsFragB(lb, 48 + rlane, hi16);
      v16h a0 = loadA16(arow0, kb, hiK);
      v16h a1 = loadA16(arow1, kb, hiK);
      c[0][0] = __builtin_amdgcn_wmma_f32_16x16x32_f16(false, a0, false, b0, (short)0, c[0][0], false, false);
      c[0][1] = __builtin_amdgcn_wmma_f32_16x16x32_f16(false, a0, false, b1, (short)0, c[0][1], false, false);
      c[0][2] = __builtin_amdgcn_wmma_f32_16x16x32_f16(false, a0, false, b2, (short)0, c[0][2], false, false);
      c[0][3] = __builtin_amdgcn_wmma_f32_16x16x32_f16(false, a0, false, b3, (short)0, c[0][3], false, false);
      c[1][0] = __builtin_amdgcn_wmma_f32_16x16x32_f16(false, a1, false, b0, (short)0, c[1][0], false, false);
      c[1][1] = __builtin_amdgcn_wmma_f32_16x16x32_f16(false, a1, false, b1, (short)0, c[1][1], false, false);
      c[1][2] = __builtin_amdgcn_wmma_f32_16x16x32_f16(false, a1, false, b2, (short)0, c[1][2], false, false);
      c[1][3] = __builtin_amdgcn_wmma_f32_16x16x32_f16(false, a1, false, b3, (short)0, c[1][3], false, false);
    }
  }

#pragma unroll
  for (int s = 0; s < 2; ++s) {
#pragma unroll
    for (int t = 0; t < 4; ++t) {
      const int col = colBase + t * 16 + rlane;
      if (col < Mstore) {
#pragma unroll
        for (int i = 0; i < 8; ++i) {
          float v = c[s][t][i];
          if (act) v = v > 0.0f ? v : 0.0f;
          const int row = rowBase + s * 16 + i + hiK;
          if (outH) outH[row * ldH + col] = (f16)v;
          if (outF) outF[row * ldF + col] = v;
        }
      }
    }
  }
}

// ---------------------------------------------------------------------------
// Dst-centric EdgeConv with mean aggregation over the 4-neighbor grid.
// out[n] = (1/deg) * sum_{nbr} relu( relu(P[n]+Q[nbr]+b1) @ W2^T + b2 )
// K-step-major: W2 fragments and (P+b1) loaded once per K-step, shared by
// all 4 neighbor accumulator sets (16 back-to-back WMMAs per K-step).
// ---------------------------------------------------------------------------
__global__ void __launch_bounds__(32) edgeconv_wmma(
    const float* __restrict__ P, const float* __restrict__ Q, int ldPQ,
    const float* __restrict__ b1,
    const f16* __restrict__ W2, int ldW2, int ksteps,
    const float* __restrict__ b2, int Mstore,
    f16* __restrict__ outH, int ldH, float* __restrict__ outF, int ldF)
{
  const int lane  = threadIdx.x;
  const int rlane = lane & 15;
  const int hiK   = (lane >> 4) << 3;
  const int hi16  = (lane >> 4) << 4;
  const int nodeBase = blockIdx.x << 4;
  const int colBase  = blockIdx.y << 6;

  // Neighbor indices for this lane's A-matrix row (node = nodeBase + lane%16).
  const int nA   = nodeBase + rlane;
  const int remA = nA & (HWIMG - 1);
  const int rA   = remA >> 7;           // / WIMG
  const int cA   = remA & (WIMG - 1);
  int nbr[4];
  nbr[0] = (rA > 0)        ? nA - WIMG : nA;
  nbr[1] = (rA < HIMG - 1) ? nA + WIMG : nA;
  nbr[2] = (cA > 0)        ? nA - 1    : nA;
  nbr[3] = (cA < WIMG - 1) ? nA + 1    : nA;

  // Existence masks + 1/degree for this lane's C-matrix rows.
  int ebits[8];
  float cinv[8];
#pragma unroll
  for (int i = 0; i < 8; ++i) {
    const int nC  = nodeBase + i + hiK;
    const int rem = nC & (HWIMG - 1);
    const int r = rem >> 7;
    const int c = rem & (WIMG - 1);
    const int e = (r > 0 ? 1 : 0) | (r < HIMG - 1 ? 2 : 0) |
                  (c > 0 ? 4 : 0) | (c < WIMG - 1 ? 8 : 0);
    ebits[i] = e;
    cinv[i]  = 1.0f / (float)__popc(e);
  }

  // Per-neighbor accumulator sets, all live across the K loop.
  v8f c2[4][4];
#pragma unroll
  for (int t = 0; t < 4; ++t) {
    const float bv = b2[colBase + t * 16 + rlane];
#pragma unroll
    for (int j = 0; j < 4; ++j)
#pragma unroll
      for (int i = 0; i < 8; ++i) c2[j][t][i] = bv;
  }

  const float* Prow = P + nA * ldPQ;
  const float* Qrow0 = Q + nbr[0] * ldPQ;
  const float* Qrow1 = Q + nbr[1] * ldPQ;
  const float* Qrow2 = Q + nbr[2] * ldPQ;
  const float* Qrow3 = Q + nbr[3] * ldPQ;

  for (int ks = 0; ks < ksteps; ++ks) {
    const int kb   = ks << 5;
    const int off0 = kb + hiK;        // first 8-value run
    const int off1 = kb + 16 + hiK;   // second 8-value run

    // Hoisted W2 fragments for this K-step.
    v16h b0 = loadB16(W2 + (colBase + 0  + rlane) * ldW2, kb, hi16);
    v16h b1f = loadB16(W2 + (colBase + 16 + rlane) * ldW2, kb, hi16);
    v16h b2f = loadB16(W2 + (colBase + 32 + rlane) * ldW2, kb, hi16);
    v16h b3f = loadB16(W2 + (colBase + 48 + rlane) * ldW2, kb, hi16);

    // Hoisted s = P + b1 for this K-step (shared across neighbors).
    v4f s0a = *(const v4f*)(Prow + off0)     + *(const v4f*)(b1 + off0);
    v4f s0b = *(const v4f*)(Prow + off0 + 4) + *(const v4f*)(b1 + off0 + 4);
    v4f s1a = *(const v4f*)(Prow + off1)     + *(const v4f*)(b1 + off1);
    v4f s1b = *(const v4f*)(Prow + off1 + 4) + *(const v4f*)(b1 + off1 + 4);

    const float* Qr[4] = {Qrow0, Qrow1, Qrow2, Qrow3};
#pragma unroll
    for (int j = 0; j < 4; ++j) {
      v4f q0a = *(const v4f*)(Qr[j] + off0);
      v4f q0b = *(const v4f*)(Qr[j] + off0 + 4);
      v4f q1a = *(const v4f*)(Qr[j] + off1);
      v4f q1b = *(const v4f*)(Qr[j] + off1 + 4);
      v16h a;
#pragma unroll
      for (int i = 0; i < 4; ++i) {
        float x0 = s0a[i] + q0a[i]; x0 = x0 > 0.0f ? x0 : 0.0f;
        float x1 = s0b[i] + q0b[i]; x1 = x1 > 0.0f ? x1 : 0.0f;
        float x2 = s1a[i] + q1a[i]; x2 = x2 > 0.0f ? x2 : 0.0f;
        float x3 = s1b[i] + q1b[i]; x3 = x3 > 0.0f ? x3 : 0.0f;
        a[i]      = (_Float16)x0;
        a[i + 4]  = (_Float16)x1;
        a[i + 8]  = (_Float16)x2;
        a[i + 12] = (_Float16)x3;
      }
      c2[j][0] = __builtin_amdgcn_wmma_f32_16x16x32_f16(false, a, false, b0,  (short)0, c2[j][0], false, false);
      c2[j][1] = __builtin_amdgcn_wmma_f32_16x16x32_f16(false, a, false, b1f, (short)0, c2[j][1], false, false);
      c2[j][2] = __builtin_amdgcn_wmma_f32_16x16x32_f16(false, a, false, b2f, (short)0, c2[j][2], false, false);
      c2[j][3] = __builtin_amdgcn_wmma_f32_16x16x32_f16(false, a, false, b3f, (short)0, c2[j][3], false, false);
    }
  }

  // relu each edge output, mask fake edges, mean, store.
  v8f acc[4];
#pragma unroll
  for (int t = 0; t < 4; ++t)
#pragma unroll
    for (int i = 0; i < 8; ++i) acc[t][i] = 0.0f;
#pragma unroll
  for (int j = 0; j < 4; ++j)
#pragma unroll
    for (int t = 0; t < 4; ++t)
#pragma unroll
      for (int i = 0; i < 8; ++i) {
        float v = c2[j][t][i];
        v = v > 0.0f ? v : 0.0f;
        if ((ebits[i] >> j) & 1) acc[t][i] += v;
      }

#pragma unroll
  for (int t = 0; t < 4; ++t) {
    const int col = colBase + t * 16 + rlane;
    if (col < Mstore) {
#pragma unroll
      for (int i = 0; i < 8; ++i) {
        const float v = acc[t][i] * cinv[i];
        const int row = nodeBase + i + hiK;
        if (outH) outH[row * ldH + col] = (f16)v;
        if (outF) outF[row * ldF + col] = v;
      }
    }
  }
}

// ---------------------------------------------------------------------------
// Prep / pack / epilogue kernels
// ---------------------------------------------------------------------------
__global__ void prep_w(f16* __restrict__ dst, const float* __restrict__ src,
                       int M, int K, int Mpad, int Kpad, int stride, int off1, int off2) {
  const int idx = blockIdx.x * 256 + threadIdx.x;
  if (idx >= Mpad * Kpad) return;
  const int m = idx / Kpad;
  const int k = idx - m * Kpad;
  float v = 0.0f;
  if (m < M && k < K) {
    v = src[m * stride + off1 + k];
    if (off2 >= 0) v -= src[m * stride + off2 + k];
  }
  dst[idx] = (f16)v;
}

__global__ void prep_b(float* __restrict__ dst, const float* __restrict__ src, int n, int npad) {
  const int i = blockIdx.x * 256 + threadIdx.x;
  if (i < npad) dst[i] = (i < n) ? src[i] : 0.0f;
}

__global__ void pack_in(const float* __restrict__ x, f16* __restrict__ Axyz, f16* __restrict__ Adir) {
  const int idx = blockIdx.x * 256 + threadIdx.x;
  if (idx >= NNODES * 96) return;
  const int n = idx / 96;
  const int k = idx - n * 96;
  const int b = n / HWIMG;
  const int rc = n & (HWIMG - 1);
  if (k < 64) {
    const float v = (k < 63) ? x[(b * CIN + k) * HWIMG + rc] : 0.0f;
    Axyz[n * 64 + k] = (f16)v;
  } else {
    const int kk = k - 64;
    const float v = (kk < 27) ? x[(b * CIN + 63 + kk) * HWIMG + rc] : 0.0f;
    Adir[n * 32 + kk] = (f16)v;
  }
}

__global__ void assemble_out(const float* __restrict__ rgb4, const float* __restrict__ sig,
                             float* __restrict__ out) {
  const int idx = blockIdx.x * 256 + threadIdx.x;
  if (idx >= NNODES * 4) return;
  const int n = idx >> 2;
  const int ch = idx & 3;
  const int b = n / HWIMG;
  const int rc = n & (HWIMG - 1);
  float v;
  if (ch < 3) {
    const float xv = rgb4[n * 4 + ch];
    v = 1.0f / (1.0f + __expf(-xv));
  } else {
    v = sig[n];
  }
  out[(b * 4 + ch) * HWIMG + rc] = v;
}

// ---------------------------------------------------------------------------
// Host side
// ---------------------------------------------------------------------------
extern "C" void kernel_launch(void* const* d_in, const int* in_sizes, int n_in,
                              void* d_out, int out_size, void* d_ws, size_t ws_size,
                              hipStream_t stream) {
  (void)n_in; (void)out_size; (void)ws_size;

  // ---- Input mapping (auto-detect flatten order) ----
  const float *Wx[8], *Bx[8];
  const float *Wf, *Bf, *Wsg, *Bsg;
  const float *E1w1, *E1b1, *E1w2, *E1b2, *E2w1, *E2b1, *E2w2, *E2b2, *Rw1, *Rb1, *Rw2, *Rb2;
  const float *X;
  if (in_sizes[0] == 2 * CIN * HWIMG) {            // insertion order: x first
    X = (const float*)d_in[0];
    int p = 1;
    for (int i = 0; i < 8; ++i) { Wx[i] = (const float*)d_in[p++]; Bx[i] = (const float*)d_in[p++]; }
    Wf  = (const float*)d_in[p++]; Bf  = (const float*)d_in[p++];
    Wsg = (const float*)d_in[p++]; Bsg = (const float*)d_in[p++];
    E1w1 = (const float*)d_in[p++]; E1b1 = (const float*)d_in[p++];
    E1w2 = (const float*)d_in[p++]; E1b2 = (const float*)d_in[p++];
    E2w1 = (const float*)d_in[p++]; E2b1 = (const float*)d_in[p++];
    E2w2 = (const float*)d_in[p++]; E2b2 = (const float*)d_in[p++];
    Rw1 = (const float*)d_in[p++]; Rb1 = (const float*)d_in[p++];
    Rw2 = (const float*)d_in[p++]; Rb2 = (const float*)d_in[p++];
  } else {                                          // JAX sorted-leaves order: x last
    E1w1 = (const float*)d_in[0];  E1b1 = (const float*)d_in[1];
    E1w2 = (const float*)d_in[2];  E1b2 = (const float*)d_in[3];
    E2w1 = (const float*)d_in[4];  E2b1 = (const float*)d_in[5];
    E2w2 = (const float*)d_in[6];  E2b2 = (const float*)d_in[7];
    Rw1  = (const float*)d_in[8];  Rb1  = (const float*)d_in[9];
    Rw2  = (const float*)d_in[10]; Rb2  = (const float*)d_in[11];
    Wsg  = (const float*)d_in[12]; Bsg  = (const float*)d_in[13];
    for (int i = 0; i < 8; ++i) { Wx[i] = (const float*)d_in[14 + 2*i]; Bx[i] = (const float*)d_in[15 + 2*i]; }
    Wf = (const float*)d_in[30]; Bf = (const float*)d_in[31];
    X  = (const float*)d_in[32];
  }

  // ---- Workspace layout (~75 MB) ----
  char* ws = (char*)d_ws;
  const size_t OFF_W   = 0;                                   // f16 weights (<2MB)
  const size_t OFF_B   = (size_t)2 << 20;                     // f32 biases
  const size_t OFF_AX  = (size_t)4 << 20;                     // N x 64 f16
  const size_t OFF_AD  = OFF_AX + (size_t)NNODES * 64 * 2;    // N x 32 f16
  const size_t OFF_HA  = OFF_AD + (size_t)NNODES * 32 * 2;    // N x 256 f16
  const size_t OFF_HB  = OFF_HA + (size_t)NNODES * 256 * 2;   // N x 256 f16
  const size_t OFF_P   = OFF_HB + (size_t)NNODES * 256 * 2;   // N x 128 f32
  const size_t OFF_Q   = OFF_P  + (size_t)NNODES * 128 * 4;   // N x 128 f32
  const size_t OFF_SIG = OFF_Q  + (size_t)NNODES * 128 * 4;   // N f32
  const size_t OFF_RGB = OFF_SIG + (size_t)NNODES * 4;        // N x 4 f32

  f16*   wbase = (f16*)(ws + OFF_W);   size_t wcur = 0;
  float* bbase = (float*)(ws + OFF_B); size_t bcur = 0;
  auto allocW = [&](int mp, int kp) { f16* p = wbase + wcur; wcur += (size_t)mp * kp; return p; };
  auto allocB = [&](int np)         { float* p = bbase + bcur; bcur += (size_t)np;    return p; };
  auto prepW = [&](f16* dst, const float* src, int M, int K, int Mpad, int Kpad,
                   int stride, int off1, int off2) {
    const int tot = Mpad * Kpad;
    prep_w<<<(tot + 255) / 256, 256, 0, stream>>>(dst, src, M, K, Mpad, Kpad, stride, off1, off2);
  };
  auto prepB = [&](float* dst, const float* src, int nv, int npad) {
    prep_b<<<(npad + 255) / 256, 256, 0, stream>>>(dst, src, nv, npad);
  };

  // ---- Weight / bias staging (f16, zero-padded) ----
  f16* W0g  = allocW(256, 64);   prepW(W0g,  Wx[0], 256, 63,  256, 64,  63,  0,   -1);
  f16* WLg[8] = {};
  for (int i = 1; i <= 7; ++i) if (i != 4) { WLg[i] = allocW(256, 256); prepW(WLg[i], Wx[i], 256, 256, 256, 256, 256, 0, -1); }
  f16* W4ag = allocW(256, 64);   prepW(W4ag, Wx[4], 256, 63,  256, 64,  319, 0,   -1);  // xyz part of skip
  f16* W4bg = allocW(256, 256);  prepW(W4bg, Wx[4], 256, 256, 256, 256, 319, 63,  -1);  // hidden part
  f16* Wfing= allocW(256, 256);  prepW(Wfing, Wf,   256, 256, 256, 256, 256, 0,   -1);
  f16* Wsigg= allocW(64, 256);   prepW(Wsigg, Wsg,  1,   256, 64,  256, 256, 0,   -1);
  // ec1: w1 (128 x 566), split at 283; feat part k<256, dir part k<27
  f16* Wd1f = allocW(128, 256);  prepW(Wd1f, E1w1, 128, 256, 128, 256, 566, 0,   283); // w1a - w1b (feat)
  f16* Wd1d = allocW(128, 32);   prepW(Wd1d, E1w1, 128, 27,  128, 32,  566, 256, 539); // w1a - w1b (dir)
  f16* Wb1f = allocW(128, 256);  prepW(Wb1f, E1w1, 128, 256, 128, 256, 566, 283, -1);  // w1b (feat)
  f16* Wb1d = allocW(128, 32);   prepW(Wb1d, E1w1, 128, 27,  128, 32,  566, 539, -1);  // w1b (dir)
  f16* W2e1 = allocW(128, 128);  prepW(W2e1, E1w2, 128, 128, 128, 128, 128, 0,   -1);
  // ec2: w1 (128 x 256), split at 128
  f16* Wd2g = allocW(128, 128);  prepW(Wd2g, E2w1, 128, 128, 128, 128, 256, 0,   128);
  f16* Wb2g = allocW(128, 128);  prepW(Wb2g, E2w1, 128, 128, 128, 128, 256, 128, -1);
  f16* W2e2 = allocW(128, 128);  prepW(W2e2, E2w2, 128, 128, 128, 128, 128, 0,   -1);
  // rgb: w1 (3 x 256) split at 128, w2 (3 x 3)
  f16* Wdrg = allocW(64, 128);   prepW(Wdrg, Rw1, 3, 128, 64, 128, 256, 0,   128);
  f16* Wbrg = allocW(64, 128);   prepW(Wbrg, Rw1, 3, 128, 64, 128, 256, 128, -1);
  f16* W2rg = allocW(64, 32);    prepW(W2rg, Rw2, 3, 3,   64, 32,  3,   0,   -1);

  float* BLg[8];
  for (int i = 0; i < 8; ++i) { BLg[i] = allocB(256); prepB(BLg[i], Bx[i], 256, 256); }
  float* Bfing = allocB(256); prepB(Bfing, Bf, 256, 256);
  float* Bsigg = allocB(64);  prepB(Bsigg, Bsg, 1, 64);
  float* B1e1g = allocB(128); prepB(B1e1g, E1b1, 128, 128);
  float* B2e1g = allocB(128); prepB(B2e1g, E1b2, 128, 128);
  float* B1e2g = allocB(128); prepB(B1e2g, E2b1, 128, 128);
  float* B2e2g = allocB(128); prepB(B2e2g, E2b2, 128, 128);
  float* B1rg  = allocB(64);  prepB(B1rg, Rb1, 3, 64);
  float* B2rg  = allocB(64);  prepB(B2rg, Rb2, 3, 64);

  f16*   AX  = (f16*)(ws + OFF_AX);
  f16*   AD  = (f16*)(ws + OFF_AD);
  f16*   HA  = (f16*)(ws + OFF_HA);
  f16*   HB  = (f16*)(ws + OFF_HB);
  float* PB  = (float*)(ws + OFF_P);
  float* QB  = (float*)(ws + OFF_Q);
  float* SIG = (float*)(ws + OFF_SIG);
  float* RGBB= (float*)(ws + OFF_RGB);

  pack_in<<<(NNODES * 96 + 255) / 256, 256, 0, stream>>>(X, AX, AD);

  auto G = [&](const f16* A1, const f16* W1, int k1, int lA1, int lW1,
               const f16* A2, const f16* W2, int k2, int lA2, int lW2,
               const float* bias, int Mstore, int gy, int act,
               f16* oH, int lH, float* oF, int lF) {
    gemm2_wmma<<<dim3(NNODES / 128, gy), 128, 0, stream>>>(
        A1, W1, k1, lA1, lW1, A2, W2, k2, lA2, lW2, bias, Mstore, act, oH, lH, oF, lF);
  };
  auto EC = [&](const float* Pp, const float* Qp, int ldpq, const float* bb1,
                const f16* w2, int ldw2, int ks, const float* bb2, int Mstore, int gy,
                f16* oH, int lH, float* oF, int lF) {
    edgeconv_wmma<<<dim3(NNODES / 16, gy), 32, 0, stream>>>(
        Pp, Qp, ldpq, bb1, w2, ldw2, ks, bb2, Mstore, oH, lH, oF, lF);
  };

  // ---- MLP trunk (ping-pong HA/HB) ----
  G(AX, W0g,    2, 64,  64,  nullptr, nullptr, 0, 0,   0,   BLg[0], 256, 4, 1, HA, 256, nullptr, 0);
  G(HA, WLg[1], 8, 256, 256, nullptr, nullptr, 0, 0,   0,   BLg[1], 256, 4, 1, HB, 256, nullptr, 0);
  G(HB, WLg[2], 8, 256, 256, nullptr, nullptr, 0, 0,   0,   BLg[2], 256, 4, 1, HA, 256, nullptr, 0);
  G(HA, WLg[3], 8, 256, 256, nullptr, nullptr, 0, 0,   0,   BLg[3], 256, 4, 1, HB, 256, nullptr, 0);
  G(AX, W4ag,   2, 64,  64,  HB,      W4bg,    8, 256, 256, BLg[4], 256, 4, 1, HA, 256, nullptr, 0); // skip layer
  G(HA, WLg[5], 8, 256, 256, nullptr, nullptr, 0, 0,   0,   BLg[5], 256, 4, 1, HB, 256, nullptr, 0);
  G(HB, WLg[6], 8, 256, 256, nullptr, nullptr, 0, 0,   0,   BLg[6], 256, 4, 1, HA, 256, nullptr, 0);
  G(HA, WLg[7], 8, 256, 256, nullptr, nullptr, 0, 0,   0,   BLg[7], 256, 4, 1, HB, 256, nullptr, 0);
  // heads: sigma (no act, f32) and feat (no act, f16)
  G(HB, Wsigg,  8, 256, 256, nullptr, nullptr, 0, 0,   0,   Bsigg,  1,   1, 0, nullptr, 0, SIG, 1);
  G(HB, Wfing,  8, 256, 256, nullptr, nullptr, 0, 0,   0,   Bfing,  256, 4, 0, HA, 256, nullptr, 0);

  // ---- EdgeConv 1: P = d@(w1a-w1b)^T, Q = d@w1b^T with d = [feat, dir] ----
  G(HA, Wd1f, 8, 256, 256, AD, Wd1d, 1, 32, 32, nullptr, 128, 2, 0, nullptr, 0, PB, 128);
  G(HA, Wb1f, 8, 256, 256, AD, Wb1d, 1, 32, 32, nullptr, 128, 2, 0, nullptr, 0, QB, 128);
  f16* E1 = HA;  // reuse (feat no longer needed)
  EC(PB, QB, 128, B1e1g, W2e1, 128, 4, B2e1g, 128, 2, E1, 128, nullptr, 0);

  // ---- EdgeConv 2 ----
  G(E1, Wd2g, 4, 128, 128, nullptr, nullptr, 0, 0, 0, nullptr, 128, 2, 0, nullptr, 0, PB, 128);
  G(E1, Wb2g, 4, 128, 128, nullptr, nullptr, 0, 0, 0, nullptr, 128, 2, 0, nullptr, 0, QB, 128);
  f16* E2 = HB;  // reuse (hcur no longer needed)
  EC(PB, QB, 128, B1e2g, W2e2, 128, 4, B2e2g, 128, 2, E2, 128, nullptr, 0);

  // ---- rgb EdgeConv (3-wide, padded) ----
  G(E2, Wdrg, 4, 128, 128, nullptr, nullptr, 0, 0, 0, nullptr, 64, 1, 0, nullptr, 0, PB, 64);
  G(E2, Wbrg, 4, 128, 128, nullptr, nullptr, 0, 0, 0, nullptr, 64, 1, 0, nullptr, 0, QB, 64);
  EC(PB, QB, 64, B1rg, W2rg, 32, 1, B2rg, 3, 1, nullptr, 0, RGBB, 4);

  // ---- sigmoid(rgb) + sigma -> (B,4,H,W) ----
  assemble_out<<<(NNODES * 4 + 255) / 256, 256, 0, stream>>>(RGBB, SIG, (float*)d_out);
}
